// MultiHeadAttention_53274774340184
// MI455X (gfx1250) — compile-verified, matrix operand order checked by reading, not run on hardware
//
#include <hip/hip_runtime.h>

typedef __bf16 bf16;
typedef __attribute__((ext_vector_type(16))) __bf16 v16bf;
typedef __attribute__((ext_vector_type(8)))  __bf16 v8bf;
typedef __attribute__((ext_vector_type(8)))  float  v8f;
// exact spelling the async-to-LDS builtin is declared with:
typedef __attribute__((__vector_size__(4 * sizeof(int)))) int i32x4v;

#define EMBED 768
#define E3    2304
#define NHEAD 12
#define HD    64
#define BATCH 2
#define SEQ   2048
#define MROWS (BATCH*SEQ)   // 4096

#ifndef __has_builtin
#define __has_builtin(x) 0
#endif
#if __has_builtin(__builtin_amdgcn_global_load_async_to_lds_b128) && \
    __has_builtin(__builtin_amdgcn_s_wait_asynccnt)
#define ASYNC_LDS 1
#else
#define ASYNC_LDS 0
#endif

// ---------------------------------------------------------------------------
// Load a 16(outer) x 32(K) bf16 operand fragment in CDNA5 WMMA A/B layout.
// p points at element (outer0, k0); ld = elements between consecutive "outer".
// Lanes 0-15 = outer 0..15; lane half h adds +8 to the K base; VGPR pairs
// 0..3 hold K=8h+0..7, pairs 4..7 hold K=16+8h+0..7 -> two 16B loads.
// ---------------------------------------------------------------------------
static __device__ __forceinline__ v16bf load_frag(const bf16* p, int ld) {
  const int lane  = threadIdx.x & 31;
  const int outer = lane & 15;
  const int h     = lane >> 4;
  const bf16* r = p + outer * ld + (h << 3);
  v8bf lo = *(const v8bf*)(r);        // K = 8h .. 8h+7
  v8bf hi = *(const v8bf*)(r + 16);   // K = 16+8h .. 16+8h+7
  v16bf f;
#pragma unroll
  for (int i = 0; i < 8; ++i) { f[i] = lo[i]; f[i + 8] = hi[i]; }
  return f;
}

static __device__ __forceinline__ v8f wmma_bf16(v16bf a, v16bf b, v8f c) {
  return __builtin_amdgcn_wmma_f32_16x16x32_bf16(false, a, false, b,
                                                 (short)0, c, false, false);
}

// ---------------------------------------------------------------------------
// fp32 -> bf16 conversion
// ---------------------------------------------------------------------------
__global__ void cvt_bf16(const float* __restrict__ src, bf16* __restrict__ dst,
                         int n) {
  int i = blockIdx.x * blockDim.x + threadIdx.x;
  if (i < n) dst[i] = (bf16)src[i];
}

// ---------------------------------------------------------------------------
// QKV projection: out(4096 x 2304) = X(4096x768) @ Wqkv^T, + bias.
// One wave computes a 64x64 tile (4x4 WMMA accumulators): per k-step,
// 4 A-frags + 4 B-frags feed 16 WMMAs (4x operand-traffic reduction).
// Epilogue scatters into Q/K/V (B,H,S,hd) bf16, folding 1/sqrt(hd) into Q.
// ---------------------------------------------------------------------------
__global__ __launch_bounds__(256) void qkv_gemm_kern(
    const bf16* __restrict__ xb, const bf16* __restrict__ wb,
    const float* __restrict__ bias,
    bf16* __restrict__ Qb, bf16* __restrict__ Kb, bf16* __restrict__ Vb) {
  const int NT = E3 / 64;                 // 36 N-tiles
  const int wave = threadIdx.x >> 5;
  const int tile = blockIdx.x * 8 + wave; // 64*36 = 2304 wave-tiles
  const int mt = tile / NT, nt = tile % NT;
  const int m0 = mt * 64, n0 = nt * 64;

  v8f acc[4][4] = {};
  for (int k = 0; k < EMBED; k += 32) {
    v16bf a[4], b[4];
#pragma unroll
    for (int i = 0; i < 4; ++i)
      a[i] = load_frag(xb + (size_t)(m0 + 16 * i) * EMBED + k, EMBED);
#pragma unroll
    for (int j = 0; j < 4; ++j)
      b[j] = load_frag(wb + (size_t)(n0 + 16 * j) * EMBED + k, EMBED);
#pragma unroll
    for (int i = 0; i < 4; ++i)
#pragma unroll
      for (int j = 0; j < 4; ++j)
        acc[i][j] = wmma_bf16(a[i], b[j], acc[i][j]);
  }

  const int lane = threadIdx.x & 31;
  const int nl = lane & 15;
  const int hb = (lane >> 4) << 3;        // C-layout row base offset
#pragma unroll
  for (int j = 0; j < 4; ++j) {
    const int col = n0 + 16 * j + nl;
    const int which = col / EMBED;        // 0=Q 1=K 2=V
    const int e = col % EMBED;
    const int hh = e >> 6, d = e & 63;
    const float bs = bias[col];
    const float scale = (which == 0) ? 0.125f : 1.0f;
    bf16* dst = (which == 0) ? Qb : ((which == 1) ? Kb : Vb);
#pragma unroll
    for (int i = 0; i < 4; ++i) {
#pragma unroll
      for (int r = 0; r < 8; ++r) {
        int m = m0 + 16 * i + hb + r;
        int bb = m >> 11;                 // / SEQ
        int s  = m & (SEQ - 1);
        dst[((size_t)((bb * NHEAD + hh) * SEQ) + s) * HD + d] =
            (bf16)((acc[i][j][r] + bs) * scale);
      }
    }
  }
}

// ---------------------------------------------------------------------------
// Flash attention. Block = 8 waves, 128 query rows of one (b,h).
// K tile staged to LDS via async DMA (ASYNCcnt path), V tile transposed
// into LDS, scores + PV via WMMA, online softmax via half-wave shfl_xor.
// ---------------------------------------------------------------------------
__global__ __launch_bounds__(256) void attention_kern(
    const bf16* __restrict__ Qb, const bf16* __restrict__ Kb,
    const bf16* __restrict__ Vb, bf16* __restrict__ AOb) {
  __shared__ bf16 Kt[128 * HD];        // [key][d]      16 KB
  __shared__ bf16 Vt[HD * 128];        // [d][key]      16 KB (transposed)
  __shared__ bf16 Pl[8][16 * 128];     // per-wave P    32 KB

  const int bh   = blockIdx.x >> 4;    // 0..23  (B*H)
  const int mt   = blockIdx.x & 15;
  const int wave = threadIdx.x >> 5;
  const int lane = threadIdx.x & 31;
  const int half = lane >> 4;
  const int nl   = lane & 15;

  const size_t base = (size_t)bh * SEQ * HD;
  const bf16* Q = Qb + base;
  const bf16* K = Kb + base;
  const bf16* V = Vb + base;

  const int m0 = mt * 128 + wave * 16;

  v16bf qa0 = load_frag(Q + (size_t)m0 * HD, HD);        // dims 0..31
  v16bf qa1 = load_frag(Q + (size_t)m0 * HD + 32, HD);   // dims 32..63

  v8f o0 = {}, o1 = {}, o2 = {}, o3 = {};
  float mrun[8], lrun[8];
#pragma unroll
  for (int j = 0; j < 8; ++j) { mrun[j] = -1e30f; lrun[j] = 0.0f; }

  for (int kt = 0; kt < SEQ; kt += 128) {
    // ---- stage K tile [128][64] (async DMA to LDS when available) ----
    for (int c = threadIdx.x; c < 1024; c += 256) {
      int key = c >> 3, dc = (c & 7) << 3;
#if ASYNC_LDS
      __builtin_amdgcn_global_load_async_to_lds_b128(
          (__attribute__((address_space(1))) i32x4v*)
              (unsigned long long)(K + (size_t)(kt + key) * HD + dc),
          (__attribute__((address_space(3))) i32x4v*)
              (unsigned)(unsigned long long)(Kt + key * HD + dc),
          0, 0);
#else
      *(v8bf*)(Kt + key * HD + dc) =
          *(const v8bf*)(K + (size_t)(kt + key) * HD + dc);
#endif
    }
    // ---- stage V tile transposed [64][128] ----
    for (int c = threadIdx.x; c < 1024; c += 256) {
      int key = c >> 3, dc = (c & 7) << 3;
      v8bf vv = *(const v8bf*)(V + (size_t)(kt + key) * HD + dc);
#pragma unroll
      for (int i = 0; i < 8; ++i) Vt[(dc + i) * 128 + key] = vv[i];
    }
    // ---- prefetch next tile while we compute this one ----
    if (kt + 128 < SEQ) {
      int t = threadIdx.x;
      const bf16* nx = ((t < 128) ? K : V) + (size_t)(kt + 128 + (t & 127)) * HD;
      __builtin_prefetch(nx, 0, 0);
    }
#if ASYNC_LDS
    __builtin_amdgcn_s_wait_asynccnt(0);
#endif
    __syncthreads();

    // ---- scores: 16 query rows x 128 keys ----
    v8f sc[8];
#pragma unroll
    for (int t = 0; t < 8; ++t) {
      v8f s = {};
      s = wmma_bf16(qa0, load_frag(Kt + t * 16 * HD, HD), s);
      s = wmma_bf16(qa1, load_frag(Kt + t * 16 * HD + 32, HD), s);
      sc[t] = s;
    }

    // ---- online softmax: row r = j + 8*half lives in one 16-lane half ----
#pragma unroll
    for (int j = 0; j < 8; ++j) {
      float mx = sc[0][j];
#pragma unroll
      for (int t = 1; t < 8; ++t) mx = fmaxf(mx, sc[t][j]);
#pragma unroll
      for (int o = 1; o < 16; o <<= 1) mx = fmaxf(mx, __shfl_xor(mx, o, 32));
      float mnew  = fmaxf(mrun[j], mx);
      float alpha = __expf(mrun[j] - mnew);
      float rsum = 0.0f;
#pragma unroll
      for (int t = 0; t < 8; ++t) {
        float p = __expf(sc[t][j] - mnew);
        sc[t][j] = p;
        rsum += p;
      }
#pragma unroll
      for (int o = 1; o < 16; o <<= 1) rsum += __shfl_xor(rsum, o, 32);
      lrun[j] = lrun[j] * alpha + rsum;
      mrun[j] = mnew;
      o0[j] *= alpha; o1[j] *= alpha; o2[j] *= alpha; o3[j] *= alpha;
    }

    // ---- P: C-layout -> row-major bf16 in LDS (per-wave region) ----
#pragma unroll
    for (int t = 0; t < 8; ++t)
#pragma unroll
      for (int j = 0; j < 8; ++j)
        Pl[wave][(j + 8 * half) * 128 + t * 16 + nl] = (bf16)sc[t][j];

    // ---- P(16x128) x V(128x64) ----
#pragma unroll
    for (int c = 0; c < 4; ++c) {
      v16bf pa = load_frag(&Pl[wave][c * 32], 128);
      o0 = wmma_bf16(pa, load_frag(Vt +  0 * 128 + c * 32, 128), o0);
      o1 = wmma_bf16(pa, load_frag(Vt + 16 * 128 + c * 32, 128), o1);
      o2 = wmma_bf16(pa, load_frag(Vt + 32 * 128 + c * 32, 128), o2);
      o3 = wmma_bf16(pa, load_frag(Vt + 48 * 128 + c * 32, 128), o3);
    }
    __syncthreads();
  }

  // ---- epilogue: normalize by row sum, store (B,S,E) bf16 ----
  const int b = bh / NHEAD, h = bh % NHEAD;
#pragma unroll
  for (int j = 0; j < 8; ++j) {
    int s = m0 + j + 8 * half;
    float inv = 1.0f / lrun[j];
    bf16* dst = AOb + ((size_t)(b * SEQ + s) * EMBED) + h * HD;
    dst[ 0 + nl] = (bf16)(o0[j] * inv);
    dst[16 + nl] = (bf16)(o1[j] * inv);
    dst[32 + nl] = (bf16)(o2[j] * inv);
    dst[48 + nl] = (bf16)(o3[j] * inv);
  }
}

// ---------------------------------------------------------------------------
// Output projection: out(4096x768) = AO(4096x768) @ Wout^T + bias (fp32 out)
// Same 64x64-per-wave register blocking.
// ---------------------------------------------------------------------------
__global__ __launch_bounds__(256) void out_gemm_kern(
    const bf16* __restrict__ ab, const bf16* __restrict__ wb,
    const float* __restrict__ bias, float* __restrict__ out) {
  const int NT = EMBED / 64;              // 12 N-tiles
  const int wave = threadIdx.x >> 5;
  const int tile = blockIdx.x * 8 + wave; // 64*12 = 768 wave-tiles
  const int mt = tile / NT, nt = tile % NT;
  const int m0 = mt * 64, n0 = nt * 64;

  v8f acc[4][4] = {};
  for (int k = 0; k < EMBED; k += 32) {
    v16bf a[4], b[4];
#pragma unroll
    for (int i = 0; i < 4; ++i)
      a[i] = load_frag(ab + (size_t)(m0 + 16 * i) * EMBED + k, EMBED);
#pragma unroll
    for (int j = 0; j < 4; ++j)
      b[j] = load_frag(wb + (size_t)(n0 + 16 * j) * EMBED + k, EMBED);
#pragma unroll
    for (int i = 0; i < 4; ++i)
#pragma unroll
      for (int j = 0; j < 4; ++j)
        acc[i][j] = wmma_bf16(a[i], b[j], acc[i][j]);
  }

  const int lane = threadIdx.x & 31;
  const int nl = lane & 15;
  const int hb = (lane >> 4) << 3;
#pragma unroll
  for (int j = 0; j < 4; ++j) {
    const int col = n0 + 16 * j + nl;
    const float bs = bias[col];
#pragma unroll
    for (int i = 0; i < 4; ++i)
#pragma unroll
      for (int r = 0; r < 8; ++r)
        out[(size_t)(m0 + 16 * i + hb + r) * EMBED + col] = acc[i][j][r] + bs;
  }
}

// ---------------------------------------------------------------------------
extern "C" void kernel_launch(void* const* d_in, const int* in_sizes, int n_in,
                              void* d_out, int out_size, void* d_ws,
                              size_t ws_size, hipStream_t stream) {
  (void)in_sizes; (void)n_in; (void)out_size; (void)ws_size;
  const float* x    = (const float*)d_in[0];
  const float* wqkv = (const float*)d_in[1];
  const float* bqkv = (const float*)d_in[2];
  const float* wout = (const float*)d_in[3];
  const float* bout = (const float*)d_in[4];
  float* out = (float*)d_out;

  // workspace carve-up (all bf16, ~36.2 MB total)
  bf16* xb    = (bf16*)d_ws;
  bf16* wqkvb = xb    + (size_t)MROWS * EMBED;
  bf16* woutb = wqkvb + (size_t)E3 * EMBED;
  bf16* Qb    = woutb + (size_t)EMBED * EMBED;
  bf16* Kb    = Qb    + (size_t)BATCH * NHEAD * SEQ * HD;
  bf16* Vb    = Kb    + (size_t)BATCH * NHEAD * SEQ * HD;
  bf16* AOb   = Vb    + (size_t)BATCH * NHEAD * SEQ * HD;

  cvt_bf16<<<(MROWS * EMBED + 255) / 256, 256, 0, stream>>>(x, xb,
                                                            MROWS * EMBED);
  cvt_bf16<<<(E3 * EMBED + 255) / 256, 256, 0, stream>>>(wqkv, wqkvb,
                                                         E3 * EMBED);
  cvt_bf16<<<(EMBED * EMBED + 255) / 256, 256, 0, stream>>>(wout, woutb,
                                                            EMBED * EMBED);

  // (4096/64)*(2304/64) = 2304 wave-tiles / 8 waves = 288 blocks
  qkv_gemm_kern<<<(MROWS / 64) * (E3 / 64) / 8, 256, 0, stream>>>(
      xb, wqkvb, bqkv, Qb, Kb, Vb);

  // B*H * (S/128) = 384 blocks
  attention_kern<<<BATCH * NHEAD * (SEQ / 128), 256, 0, stream>>>(Qb, Kb, Vb,
                                                                  AOb);

  // (4096/64)*(768/64) = 768 wave-tiles / 8 waves = 96 blocks
  out_gemm_kern<<<(MROWS / 64) * (EMBED / 64) / 8, 256, 0, stream>>>(
      AOb, woutb, bout, out);
}
